// MultiHeadAttentionQuantum_65481071410575
// MI455X (gfx1250) — matrix operand order repaired and checked by reading, not compiled, not run on hardware
//
#include <hip/hip_runtime.h>
#include <hip/hip_bf16.h>

typedef _Float16 half_t;
typedef _Float16 v16h __attribute__((ext_vector_type(16)));
typedef float    v8f  __attribute__((ext_vector_type(8)));

#define EMBED 1024
#define SEQ   4096
#define BATCH 4
#define SCALE 0.125f   // 1/sqrt(64)

// Padded LDS row strides (halves): +8 halves (16B) shifts each lane's bank by
// 4, so 16 lanes reading stride-ASTRIDE rows hit 16 distinct banks.
#define ASTRIDE 1032   // 1024 + 8
#define PSTRIDE 264    // 256 + 8

// ---------------------------------------------------------------------------
// Fragment loaders (contiguous-K layout, same for A and B, wave32)
// A 16x32 f16: lane L (g = L>>4, ln = L&15) holds row ln,
//   16 contiguous K at k0 + 16g.
// B 32x16 f16: lane L holds column ln, 16 contiguous K at k0 + 16g.
// ---------------------------------------------------------------------------
__device__ __forceinline__ v16h load_a_frag(const half_t* row, int k0, int g) {
    v16h a;
    uint4* ap = reinterpret_cast<uint4*>(&a);
    ap[0] = *reinterpret_cast<const uint4*>(row + k0 + 16 * g);
    ap[1] = *reinterpret_cast<const uint4*>(row + k0 + 16 * g + 8);
    return a;
}
__device__ __forceinline__ v16h load_b_frag(const half_t* row, int k0, int g) {
    v16h b;
    uint4* bp = reinterpret_cast<uint4*>(&b);
    const half_t* p = row + k0 + 16 * g;
    bp[0] = *reinterpret_cast<const uint4*>(p);
    bp[1] = *reinterpret_cast<const uint4*>(p + 8);
    return b;
}
#define WMMA_F16(a, b, c) \
    __builtin_amdgcn_wmma_f32_16x16x32_f16(false, (a), false, (b), (short)0, (c), false, false)

// ---------------------------------------------------------------------------
// Kernel 1: proj[b,s,e] = cos(theta_flat[e]) * cos(x[b,s,e])  (f32 -> f16)
// ---------------------------------------------------------------------------
__global__ __launch_bounds__(256) void prep_proj_kernel(const float* __restrict__ x,
                                                        const float* __restrict__ theta,
                                                        half_t* __restrict__ proj) {
    size_t i = (size_t)blockIdx.x * 256 + threadIdx.x;
    int e = (int)(i & (EMBED - 1));
    proj[i] = (half_t)(__cosf(theta[e]) * __cosf(x[i]));
}

__global__ __launch_bounds__(256) void prep_wc_kernel(const float* __restrict__ wc,
                                                      half_t* __restrict__ wch) {
    size_t i = (size_t)blockIdx.x * 256 + threadIdx.x;
    wch[i] = (half_t)wc[i];
}

// ---------------------------------------------------------------------------
// Kernel 2: LDS-tiled transpose  projT[b,e,s] = proj[b,s,e]  (64x64 tiles)
// ---------------------------------------------------------------------------
__global__ __launch_bounds__(256) void transpose_kernel(const half_t* __restrict__ proj,
                                                        half_t* __restrict__ projT) {
    __shared__ half_t tile[64][68];
    int b  = blockIdx.z;
    int s0 = blockIdx.x * 64;
    int e0 = blockIdx.y * 64;
    const half_t* src = proj  + (size_t)b * SEQ * EMBED;
    half_t*       dst = projT + (size_t)b * SEQ * EMBED;
    int tx = threadIdx.x & 63;   // fast dim (coalesced)
    int ty = threadIdx.x >> 6;   // 0..3
#pragma unroll
    for (int j = 0; j < 16; ++j) {
        int s = ty + j * 4;
        tile[s][tx] = src[(size_t)(s0 + s) * EMBED + e0 + tx];
    }
    __syncthreads();
#pragma unroll
    for (int j = 0; j < 16; ++j) {
        int e = ty + j * 4;
        dst[(size_t)(e0 + e) * SEQ + s0 + tx] = tile[tx][e];
    }
}

// Stage 16 rows x EMBED halves from global (row-major) into padded LDS.
__device__ __forceinline__ void stage_rows(const half_t* __restrict__ src, half_t* dst) {
    int row = threadIdx.x >> 4;        // 0..15
    int col = threadIdx.x & 15;        // 0..15 (uint4 granule)
    const uint4* s = reinterpret_cast<const uint4*>(src + (size_t)row * EMBED);
    uint4*       d = reinterpret_cast<uint4*>(dst + row * ASTRIDE);
#pragma unroll
    for (int j = 0; j < 8; ++j) d[col + j * 16] = s[col + j * 16];
}

// ---------------------------------------------------------------------------
// Kernel 3: softmax stats. One WG = 16 rows. Each wave computes two 16x16
// score tiles per 256-column sweep; one A-fragment feeds both accumulator
// chains (2x ILP, half the LDS traffic). Online (max, sum-exp) per lane,
// reduced via shuffles + LDS.
// ---------------------------------------------------------------------------
__global__ __launch_bounds__(256) void stats_kernel(const half_t* __restrict__ proj,
                                                    float* __restrict__ mOut,
                                                    float* __restrict__ linvOut) {
    __shared__ __align__(16) half_t As[16 * ASTRIDE];
    __shared__ float redM[8][16];
    __shared__ float redL[8][16];

    int b = blockIdx.y;
    int row0 = blockIdx.x * 16;
    const half_t* projb = proj + (size_t)b * SEQ * EMBED;

    stage_rows(projb + (size_t)row0 * EMBED, As);
    __syncthreads();

    int lane = threadIdx.x & 31, wave = threadIdx.x >> 5;
    int g = lane >> 4, ln = lane & 15;
    const half_t* arow = As + ln * ASTRIDE;

    float lm[8], ll[8];
#pragma unroll
    for (int i = 0; i < 8; ++i) { lm[i] = -1e30f; ll[i] = 0.f; }

    for (int t0 = 0; t0 < SEQ; t0 += 256) {
        const half_t* brow0 = projb + (size_t)(t0 + wave * 32 + ln) * EMBED;
        const half_t* brow1 = brow0 + (size_t)16 * EMBED;
        v8f acc0 = {}, acc1 = {};
#pragma unroll 8
        for (int k0 = 0; k0 < EMBED; k0 += 32) {
            v16h a = load_a_frag(arow, k0, g);
            acc0 = WMMA_F16(a, load_b_frag(brow0, k0, g), acc0);
            acc1 = WMMA_F16(a, load_b_frag(brow1, k0, g), acc1);
        }
#pragma unroll
        for (int i = 0; i < 8; ++i) {
            float s0 = acc0[i] * SCALE, s1 = acc1[i] * SCALE;
            float nm = fmaxf(lm[i], fmaxf(s0, s1));
            ll[i] = ll[i] * __expf(lm[i] - nm) + __expf(s0 - nm) + __expf(s1 - nm);
            lm[i] = nm;
        }
    }
    // reduce across the 16 lanes holding the same rows
#pragma unroll
    for (int off = 1; off < 16; off <<= 1) {
#pragma unroll
        for (int i = 0; i < 8; ++i) {
            float om = __shfl_xor(lm[i], off, 16);
            float ol = __shfl_xor(ll[i], off, 16);
            float nm = fmaxf(lm[i], om);
            ll[i] = ll[i] * __expf(lm[i] - nm) + ol * __expf(om - nm);
            lm[i] = nm;
        }
    }
    if (ln == 0) {
#pragma unroll
        for (int i = 0; i < 8; ++i) { redM[wave][g * 8 + i] = lm[i]; redL[wave][g * 8 + i] = ll[i]; }
    }
    __syncthreads();
    if (threadIdx.x < 16) {
        float m = redM[0][threadIdx.x], l = redL[0][threadIdx.x];
#pragma unroll
        for (int w = 1; w < 8; ++w) {
            float om = redM[w][threadIdx.x], ol = redL[w][threadIdx.x];
            float nm = fmaxf(m, om);
            l = l * __expf(m - nm) + ol * __expf(om - nm);
            m = nm;
        }
        mOut[(size_t)b * SEQ + row0 + threadIdx.x]    = m;
        linvOut[(size_t)b * SEQ + row0 + threadIdx.x] = 1.0f / l;
    }
}

// ---------------------------------------------------------------------------
// Kernel 4: apply. Recompute score tiles, P = exp(s-m)/l -> f16 in LDS, then
// O(16x128 per wave) += P(16x256) x V(256x128). The P-fragment is hoisted:
// one A-frag per K-chunk feeds 8 independent accumulator chains.
// ---------------------------------------------------------------------------
__global__ __launch_bounds__(256) void apply_kernel(const half_t* __restrict__ proj,
                                                    const half_t* __restrict__ projT,
                                                    const float* __restrict__ mIn,
                                                    const float* __restrict__ linvIn,
                                                    half_t* __restrict__ Oout) {
    __shared__ __align__(16) half_t As[16 * ASTRIDE];
    __shared__ __align__(16) half_t Ps[16 * PSTRIDE];

    int b = blockIdx.y;
    int row0 = blockIdx.x * 16;
    const half_t* projb  = proj  + (size_t)b * SEQ * EMBED;
    const half_t* projTb = projT + (size_t)b * SEQ * EMBED;

    stage_rows(projb + (size_t)row0 * EMBED, As);
    __syncthreads();

    int lane = threadIdx.x & 31, wave = threadIdx.x >> 5;
    int g = lane >> 4, ln = lane & 15;
    const half_t* arow = As + ln * ASTRIDE;
    const half_t* prow = Ps + ln * PSTRIDE;
    int c0 = wave * 128;
    const half_t* vbase = projTb + (size_t)(c0 + ln) * SEQ;   // + nn*16*SEQ per tile

    float mr[8], lr[8];
#pragma unroll
    for (int i = 0; i < 8; ++i) {
        int r = row0 + g * 8 + i;
        mr[i] = mIn[(size_t)b * SEQ + r];
        lr[i] = linvIn[(size_t)b * SEQ + r];
    }

    v8f o[8];
#pragma unroll
    for (int nn = 0; nn < 8; ++nn) o[nn] = {};

    for (int t0 = 0; t0 < SEQ; t0 += 256) {
        // ---- phase 1: scores -> probabilities (two tiles, shared A-frag)
        {
            int tloc = wave * 32;
            const half_t* brow0 = projb + (size_t)(t0 + tloc + ln) * EMBED;
            const half_t* brow1 = brow0 + (size_t)16 * EMBED;
            v8f acc0 = {}, acc1 = {};
#pragma unroll 8
            for (int k0 = 0; k0 < EMBED; k0 += 32) {
                v16h a = load_a_frag(arow, k0, g);
                acc0 = WMMA_F16(a, load_b_frag(brow0, k0, g), acc0);
                acc1 = WMMA_F16(a, load_b_frag(brow1, k0, g), acc1);
            }
#pragma unroll
            for (int i = 0; i < 8; ++i) {
                float p0 = __expf(acc0[i] * SCALE - mr[i]) * lr[i];
                float p1 = __expf(acc1[i] * SCALE - mr[i]) * lr[i];
                Ps[(g * 8 + i) * PSTRIDE + tloc + ln]      = (half_t)p0;
                Ps[(g * 8 + i) * PSTRIDE + tloc + 16 + ln] = (half_t)p1;
            }
        }
        __syncthreads();
        // ---- phase 2: K outermost; one P-frag -> 8 independent chains
#pragma unroll
        for (int k0 = 0; k0 < 256; k0 += 32) {
            v16h a = load_a_frag(prow, k0, g);
            const half_t* v0 = vbase + t0;
#pragma unroll
            for (int nn = 0; nn < 8; ++nn)
                o[nn] = WMMA_F16(a, load_b_frag(v0 + (size_t)nn * 16 * SEQ, k0, g), o[nn]);
        }
        __syncthreads();
    }
    // store attention output as f16 for the final GEMM
#pragma unroll
    for (int nn = 0; nn < 8; ++nn)
#pragma unroll
        for (int i = 0; i < 8; ++i)
            Oout[((size_t)b * SEQ + row0 + g * 8 + i) * EMBED + c0 + nn * 16 + ln] = (half_t)o[nn][i];
}

// ---------------------------------------------------------------------------
// Kernel 5: out = O @ W_c^T + b_c. Two 16x16 tiles per wave share each
// A-fragment (2 independent accumulator chains).
// ---------------------------------------------------------------------------
__global__ __launch_bounds__(256) void out_gemm_kernel(const half_t* __restrict__ Oin,
                                                       const half_t* __restrict__ Wch,
                                                       const float* __restrict__ bc,
                                                       float* __restrict__ out) {
    __shared__ __align__(16) half_t As[16 * ASTRIDE];
    size_t row0 = (size_t)blockIdx.x * 16;            // over M = B*S
    stage_rows(Oin + row0 * EMBED, As);
    __syncthreads();

    int lane = threadIdx.x & 31, wave = threadIdx.x >> 5;
    int g = lane >> 4, ln = lane & 15;
    int c0 = blockIdx.y * 256 + wave * 32;
    const half_t* arow  = As + ln * ASTRIDE;
    const half_t* brow0 = Wch + (size_t)(c0 + ln) * EMBED;
    const half_t* brow1 = brow0 + (size_t)16 * EMBED;

    v8f acc0 = {}, acc1 = {};
#pragma unroll 8
    for (int k0 = 0; k0 < EMBED; k0 += 32) {
        v16h a = load_a_frag(arow, k0, g);
        acc0 = WMMA_F16(a, load_b_frag(brow0, k0, g), acc0);
        acc1 = WMMA_F16(a, load_b_frag(brow1, k0, g), acc1);
    }

    float bv0 = bc[c0 + ln];
    float bv1 = bc[c0 + 16 + ln];
#pragma unroll
    for (int i = 0; i < 8; ++i) {
        out[(row0 + g * 8 + i) * EMBED + c0 + ln]      = acc0[i] + bv0;
        out[(row0 + g * 8 + i) * EMBED + c0 + 16 + ln] = acc1[i] + bv1;
    }
}

// ---------------------------------------------------------------------------
extern "C" void kernel_launch(void* const* d_in, const int* in_sizes, int n_in,
                              void* d_out, int out_size, void* d_ws, size_t ws_size,
                              hipStream_t stream) {
    const float* x     = (const float*)d_in[0];   // (B,S,E)
    const float* theta = (const float*)d_in[1];   // (H,D) == flat E
    const float* W_c   = (const float*)d_in[2];   // (E,E)
    const float* b_c   = (const float*)d_in[3];   // (E)
    float* out = (float*)d_out;

    const size_t PROJ_BYTES = (size_t)BATCH * SEQ * EMBED * sizeof(half_t); // 33.5 MB
    char* ws = (char*)d_ws;
    half_t* proj  = (half_t*)(ws);
    half_t* projT = (half_t*)(ws + PROJ_BYTES);
    half_t* Oh    = (half_t*)(ws + 2 * PROJ_BYTES);
    half_t* Wch   = (half_t*)(ws + 3 * PROJ_BYTES);
    float*  mBuf  = (float*)(ws + 3 * PROJ_BYTES + (size_t)EMBED * EMBED * sizeof(half_t));
    float*  lBuf  = mBuf + (size_t)BATCH * SEQ;

    const size_t total = (size_t)BATCH * SEQ * EMBED;

    prep_proj_kernel<<<dim3(total / 256), dim3(256), 0, stream>>>(x, theta, proj);
    prep_wc_kernel<<<dim3((EMBED * EMBED) / 256), dim3(256), 0, stream>>>(W_c, Wch);
    transpose_kernel<<<dim3(SEQ / 64, EMBED / 64, BATCH), dim3(256), 0, stream>>>(proj, projT);
    stats_kernel<<<dim3(SEQ / 16, BATCH), dim3(256), 0, stream>>>(proj, mBuf, lBuf);
    apply_kernel<<<dim3(SEQ / 16, BATCH), dim3(256), 0, stream>>>(proj, projT, mBuf, lBuf, Oh);
    out_gemm_kernel<<<dim3((BATCH * SEQ) / 16, EMBED / 256), dim3(256), 0, stream>>>(Oh, Wch, b_c, out);
}
